// MultiDetector_22110491639962
// MI455X (gfx1250) — matrix operand order repaired and checked by reading, not compiled, not run on hardware
//
#include <hip/hip_runtime.h>

// ---------------------------------------------------------------------------
// MI455X (gfx1250) implementation: whole network lowered to bf16 WMMA GEMMs.
// Activations kept channel-last [b, p, c]; heads become GEMMs with strided
// row addressing; dilated 3x3x3 convs go through im2col.
// GEMM uses double-buffered LDS fed by GLOBAL_LOAD_ASYNC_TO_LDS_B128
// (ASYNCcnt-tracked) when the toolchain exposes the builtin.
// ---------------------------------------------------------------------------

typedef __attribute__((ext_vector_type(16))) __bf16 v16bf;
typedef __attribute__((ext_vector_type(8)))  __bf16 v8bf;
typedef __attribute__((ext_vector_type(8)))  float  v8f;
typedef int i4v __attribute__((vector_size(16)));   // int4 vector for async builtin

#if defined(__AMDGCN__) && \
    __has_builtin(__builtin_amdgcn_global_load_async_to_lds_b128) && \
    __has_builtin(__builtin_amdgcn_s_wait_asynccnt)
#define USE_ASYNC_LDS 1
#else
#define USE_ASYNC_LDS 0
#endif

__device__ __forceinline__ unsigned short f2bf(float f) {
  unsigned int u = __builtin_bit_cast(unsigned int, f);
  u += 0x7FFFu + ((u >> 16) & 1u);   // round-to-nearest-even
  return (unsigned short)(u >> 16);
}

// 16B global -> LDS copy: async (no VGPR round trip) when available.
__device__ __forceinline__ void cp16_g2l(const unsigned short* g,
                                         unsigned short* l) {
#if USE_ASYNC_LDS
  __builtin_amdgcn_global_load_async_to_lds_b128(
      (__attribute__((address_space(1))) i4v*)g,
      (__attribute__((address_space(3))) i4v*)l, 0, 0);
#else
  *(uint4*)l = *(const uint4*)g;
#endif
}

__device__ __forceinline__ void wait_g2l() {
#if USE_ASYNC_LDS
  __builtin_amdgcn_s_wait_asynccnt(0);
#endif
}

union ABFrag {
  v16bf whole;
  v8bf  half[2];
};

#define BM   128
#define BN   128
#define BK   32
#define LDSW 40   // bf16 per LDS row: 80B stride, 16B aligned, bank-conflict free

// C[M,N] (+ optional bf16 store) = A x B^T, B given as [N,K] row-major bf16.
// A row m starts at A + (m/dOut)*batchStride + (m%dOut)*rowStep  (elements).
// Requires: K % 32 == 0, M % 128 == 0 (true for all launches here).
__global__ __launch_bounds__(256, 1) void wmma_gemm_bf16(
    const unsigned short* __restrict__ A,
    const unsigned short* __restrict__ Bw,
    void* __restrict__ C,
    int N, int K,
    int dOut, long long batchStride, long long rowStep,
    int ldc, int storeBf16)
{
  __shared__ __align__(16) unsigned short lA[2][BM * LDSW];
  __shared__ __align__(16) unsigned short lB[2][BN * LDSW];

  const int tid   = threadIdx.x;
  const int lane  = tid & 31;
  const int wave  = tid >> 5;
  const int wm    = wave & 3;   // 4 M-waves
  const int wn    = wave >> 2;  // 2 N-waves
  const int lrow  = lane & 15;
  const int khalf = lane >> 4;

  const int mBlock = blockIdx.y * BM;
  const int nBlock = blockIdx.x * BN;

  // --- global -> LDS staging descriptors (16B chunk per thread per half) ---
  const unsigned short* aSrc[2];
  const unsigned short* bSrc[2];
  int  aDst[2], bDst[2];
  bool bOk[2];
#pragma unroll
  for (int s = 0; s < 2; ++s) {
    int chunk = tid + s * 256;          // 512 chunks of 8 bf16 per tile
    int row = chunk >> 2;
    int q   = chunk & 3;
    int gm  = mBlock + row;
    long long base = (long long)(gm / dOut) * batchStride
                   + (long long)(gm % dOut) * rowStep + q * 8;
    aSrc[s] = A + base;
    aDst[s] = row * LDSW + q * 8;
    int gn  = nBlock + row;
    bOk[s]  = (gn < N);
    bSrc[s] = Bw + (long long)gn * K + q * 8;
    bDst[s] = row * LDSW + q * 8;
  }

  // --- LDS fragment read offsets (ISA 16-bit A 16x32 / B 32x16 layouts) ---
  int aOff[2], bOff[4];
#pragma unroll
  for (int mf = 0; mf < 2; ++mf)
    aOff[mf] = (wm * 32 + mf * 16 + lrow) * LDSW + khalf * 8;
#pragma unroll
  for (int nf = 0; nf < 4; ++nf)
    bOff[nf] = (wn * 64 + nf * 16 + lrow) * LDSW + khalf * 8;

  v8f acc[2][4];
#pragma unroll
  for (int mf = 0; mf < 2; ++mf)
#pragma unroll
    for (int nf = 0; nf < 4; ++nf)
      acc[mf][nf] = v8f{0.f,0.f,0.f,0.f,0.f,0.f,0.f,0.f};

  // prologue: fetch tile 0 into buffer 0
#pragma unroll
  for (int s = 0; s < 2; ++s) {
    cp16_g2l(aSrc[s], &lA[0][aDst[s]]);
    if (bOk[s]) cp16_g2l(bSrc[s], &lB[0][bDst[s]]);
  }
  // out-of-range B rows are never fetched: zero them once in both buffers
#pragma unroll
  for (int s = 0; s < 2; ++s) {
    if (!bOk[s]) {
      uint4 z; z.x = z.y = z.z = z.w = 0u;
      *(uint4*)&lB[0][bDst[s]] = z;
      *(uint4*)&lB[1][bDst[s]] = z;
    }
  }

  const int nk = K / BK;
  for (int t = 0; t < nk; ++t) {
    const int cur = t & 1;
    wait_g2l();           // my async contributions to buffer `cur` done
    __syncthreads();      // everyone's done; buffer cur^1 free to overwrite

    if (t + 1 < nk) {     // fetch tile t+1 into the other buffer
      int kt2 = (t + 1) * BK;
#pragma unroll
      for (int s = 0; s < 2; ++s) {
        cp16_g2l(aSrc[s] + kt2, &lA[cur ^ 1][aDst[s]]);
        if (bOk[s]) cp16_g2l(bSrc[s] + kt2, &lB[cur ^ 1][bDst[s]]);
      }
    }

    ABFrag af[2];
#pragma unroll
    for (int mf = 0; mf < 2; ++mf) {
      af[mf].half[0] = *(const v8bf*)&lA[cur][aOff[mf]];        // K=kh*8..+7
      af[mf].half[1] = *(const v8bf*)&lA[cur][aOff[mf] + 16];   // K=16+kh*8..+7
    }
    ABFrag bf[4];
#pragma unroll
    for (int nf = 0; nf < 4; ++nf) {
      bf[nf].half[0] = *(const v8bf*)&lB[cur][bOff[nf]];        // K=kh*8..+7
      bf[nf].half[1] = *(const v8bf*)&lB[cur][bOff[nf] + 16];   // K=16+kh*8..+7
    }
#pragma unroll
    for (int mf = 0; mf < 2; ++mf)
#pragma unroll
      for (int nf = 0; nf < 4; ++nf)
        acc[mf][nf] = __builtin_amdgcn_wmma_f32_16x16x32_bf16(
            false, af[mf].whole, false, bf[nf].whole,
            (short)0, acc[mf][nf], false, false);
  }

  // --- epilogue: C/D layout VGPR r -> M = khalf*8 + r, N = lrow ---
#pragma unroll
  for (int nf = 0; nf < 4; ++nf) {
    int col = nBlock + wn * 64 + nf * 16 + lrow;
    if (col < N) {
#pragma unroll
      for (int mf = 0; mf < 2; ++mf) {
        int rowBase = mBlock + wm * 32 + mf * 16 + khalf * 8;
#pragma unroll
        for (int r = 0; r < 8; ++r) {
          long long idx = (long long)(rowBase + r) * ldc + col;
          float v = acc[mf][nf][r];
          if (storeBf16) ((unsigned short*)C)[idx] = f2bf(v);
          else           ((float*)C)[idx] = v;
        }
      }
    }
  }
}

// x NCDHW [128,2048,16,2,2] f32 -> channel-last bf16 [b][p=64][c=2048]
__global__ void convert_x_k(const float* __restrict__ x,
                            unsigned short* __restrict__ xt)
{
  __shared__ float tile[64][65];
  int b  = blockIdx.y;
  int c0 = blockIdx.x * 64;
  const float* src = x + ((long long)b * 2048 + c0) * 64;
  for (int i = threadIdx.x; i < 64 * 64; i += 256) {
    int cl = i >> 6, p = i & 63;
    tile[cl][p] = src[cl * 64 + p];
  }
  __syncthreads();
  unsigned short* dst = xt + (long long)b * 64 * 2048 + c0;
  for (int i = threadIdx.x; i < 64 * 64; i += 256) {
    int p = i >> 6, cl = i & 63;
    dst[(long long)p * 2048 + cl] = f2bf(tile[cl][p]);
  }
}

// weights [N][C][T] f32 -> [N][T][C] bf16  (T = 1, 8 or 27)
__global__ void reorder_w_k(const float* __restrict__ in,
                            unsigned short* __restrict__ out, int C, int T)
{
  int n = blockIdx.y;
  int c = blockIdx.x * blockDim.x + threadIdx.x;
  if (c >= C) return;
  const float* src = in + ((long long)n * C + c) * T;
  float vals[27];
  for (int t = 0; t < T; ++t) vals[t] = src[t];
  for (int t = 0; t < T; ++t)
    out[((long long)n * T + t) * C + c] = f2bf(vals[t]);
}

// im2col for 3x3x3 conv, D stride/dil/pad = 2, HW pad 1.  in: [128][Din*4][C]
// out row m=(b,do,ho,wo): [tap*C + c], tap = kd*9+kh*3+kw, zero for OOB taps.
__global__ void im2col_k(const unsigned short* __restrict__ in,
                         unsigned short* __restrict__ out,
                         int Din, int Dout, int C)
{
  int m  = blockIdx.x;
  int b  = m / (Dout * 4);
  int p  = m % (Dout * 4);
  int dd = p >> 2, ho = (p >> 1) & 1, wo = p & 1;
  int chunks = C >> 3;
  int total  = 27 * chunks;
  for (int j = threadIdx.x; j < total; j += blockDim.x) {
    int tap = j / chunks;
    int cc  = (j % chunks) * 8;
    int kd = tap / 9, rr = tap % 9, kh = rr / 3, kw = rr % 3;
    int din = 2 * dd - 2 + 2 * kd;
    int hin = ho - 1 + kh;
    int win = wo - 1 + kw;
    uint4 v; v.x = v.y = v.z = v.w = 0u;
    if (din >= 0 && din < Din && hin >= 0 && hin < 2 && win >= 0 && win < 2)
      v = *(const uint4*)&in[((long long)(b * Din * 4)
            + (din * 4 + hin * 2 + win)) * C + cc];
    *(uint4*)&out[(long long)m * (27 * C) + (long long)tap * C + cc] = v;
  }
}

// head GEMM output [B*Ds][5] f32 -> loc/conf concat layout
__global__ void scatter_head_k(const float* __restrict__ hout,
                               float* __restrict__ outLoc,
                               float* __restrict__ outConf,
                               int Ds, int locBase, int confBase, int total)
{
  int i = blockIdx.x * blockDim.x + threadIdx.x;
  if (i >= total) return;
  int n = i % 5;
  int m = i / 5;
  int d = m % Ds;
  int b = m / Ds;
  float v = hout[(long long)m * 5 + n];
  if (n < 2) outLoc [(long long)b * 52 + locBase  + n * Ds + d] = v;
  else       outConf[(long long)b * 78 + confBase + (n - 2) * Ds + d] = v;
}

extern "C" void kernel_launch(void* const* d_in, const int* in_sizes, int n_in,
                              void* d_out, int out_size, void* d_ws, size_t ws_size,
                              hipStream_t stream) {
  (void)in_sizes; (void)n_in; (void)out_size; (void)ws_size;
  const float* x = (const float*)d_in[0];
  const float* loc_w[4]  = {(const float*)d_in[2], (const float*)d_in[6],
                            (const float*)d_in[10], (const float*)d_in[14]};
  const float* conf_w[4] = {(const float*)d_in[3], (const float*)d_in[7],
                            (const float*)d_in[11], (const float*)d_in[15]};
  const float* e_w1[3] = {(const float*)d_in[4], (const float*)d_in[8],
                          (const float*)d_in[12]};
  const float* e_w2[3] = {(const float*)d_in[5], (const float*)d_in[9],
                          (const float*)d_in[13]};

  const int Bn = 128;
  const int Cin[4]      = {2048, 1024, 512, 256};  // channels of each activation
  const int Mid[3]      = {512, 256, 128};
  const int Din[4]      = {16, 8, 4, 2};           // temporal size of each activation
  const int Ds[4]       = {15, 7, 3, 1};           // head temporal outputs
  const int locBase[4]  = {0, 30, 44, 50};
  const int confBase[4] = {0, 45, 66, 75};

  char* ws = (char*)d_ws;
  size_t off = 0;
  auto alloc = [&](size_t bytes) -> char* {
    char* p = ws + off;
    off += (bytes + 255) & ~(size_t)255;
    return p;
  };

  unsigned short* act[4];
  for (int s = 0; s < 4; ++s)
    act[s] = (unsigned short*)alloc((size_t)Bn * Din[s] * 4 * Cin[s] * 2);
  unsigned short* headW[4];
  for (int s = 0; s < 4; ++s)
    headW[s] = (unsigned short*)alloc((size_t)5 * 8 * Cin[s] * 2);
  unsigned short* w1t[3];
  for (int i = 0; i < 3; ++i)
    w1t[i] = (unsigned short*)alloc((size_t)Mid[i] * Cin[i] * 2);
  unsigned short* w2t[3];
  for (int i = 0; i < 3; ++i)
    w2t[i] = (unsigned short*)alloc((size_t)Cin[i + 1] * 27 * Mid[i] * 2);
  unsigned short* mid[3];
  for (int i = 0; i < 3; ++i)
    mid[i] = (unsigned short*)alloc((size_t)Bn * Din[i] * 4 * Mid[i] * 2);
  unsigned short* colBuf =
      (unsigned short*)alloc((size_t)Bn * Din[1] * 4 * 27 * Mid[0] * 2);  // max
  float* headOut[4];
  for (int s = 0; s < 4; ++s)
    headOut[s] = (float*)alloc((size_t)Bn * Ds[s] * 5 * 4);

  auto gemm = [&](const unsigned short* A, const unsigned short* Bm, void* C,
                  int M, int N, int K, int dOut, long long bs, long long rs,
                  int ldc, int storeBf16) {
    dim3 grid((N + BN - 1) / BN, M / BM);
    wmma_gemm_bf16<<<grid, 256, 0, stream>>>(A, Bm, C, N, K, dOut, bs, rs,
                                             ldc, storeBf16);
  };

  // x -> channel-last bf16
  convert_x_k<<<dim3(32, 128), 256, 0, stream>>>(x, act[0]);

  // weight reorders (heads: loc rows 0-1 + conf rows 2-4, K order [tap][c])
  for (int s = 0; s < 4; ++s) {
    dim3 g((Cin[s] + 255) / 256, 2);
    reorder_w_k<<<g, 256, 0, stream>>>(loc_w[s], headW[s], Cin[s], 8);
    dim3 g2((Cin[s] + 255) / 256, 3);
    reorder_w_k<<<g2, 256, 0, stream>>>(conf_w[s], headW[s] + 2 * 8 * Cin[s],
                                        Cin[s], 8);
  }
  for (int i = 0; i < 3; ++i) {
    dim3 g((Cin[i] + 255) / 256, Mid[i]);
    reorder_w_k<<<g, 256, 0, stream>>>(e_w1[i], w1t[i], Cin[i], 1);
    dim3 g2((Mid[i] + 255) / 256, Cin[i + 1]);
    reorder_w_k<<<g2, 256, 0, stream>>>(e_w2[i], w2t[i], Mid[i], 27);
  }

  // extra-layer chain: 1x1x1 GEMM -> im2col -> 3x3x3 GEMM
  for (int i = 0; i < 3; ++i) {
    int M1 = Bn * Din[i] * 4;
    gemm(act[i], w1t[i], mid[i], M1, Mid[i], Cin[i],
         M1, 0, (long long)Cin[i], Mid[i], 1);
    int Dout = Din[i + 1];
    int M2 = Bn * Dout * 4;
    im2col_k<<<M2, 256, 0, stream>>>(mid[i], colBuf, Din[i], Dout, Mid[i]);
    gemm(colBuf, w2t[i], act[i + 1], M2, Cin[i + 1], 27 * Mid[i],
         M2, 0, (long long)27 * Mid[i], Cin[i + 1], 1);
  }

  // heads: strided-row GEMMs (K window = 8 contiguous p-slices) + scatter
  float* outLoc  = (float*)d_out;
  float* outConf = outLoc + 128 * 52;
  for (int s = 0; s < 4; ++s) {
    int M = Bn * Ds[s];
    gemm(act[s], headW[s], headOut[s], M, 5, 8 * Cin[s],
         Ds[s], (long long)Din[s] * 4 * Cin[s], (long long)4 * Cin[s], 5, 0);
    int tot = M * 5;
    scatter_head_k<<<(tot + 255) / 256, 256, 0, stream>>>(
        headOut[s], outLoc, outConf, Ds[s], locBase[s], confBase[s], tot);
  }
}
